// Pairwise_19713899889258
// MI455X (gfx1250) — matrix operand order repaired and checked by reading, not compiled, not run on hardware
//
#include <hip/hip_runtime.h>
#include <hip/hip_bf16.h>

typedef __attribute__((ext_vector_type(16))) _Float16 v16h;
typedef __attribute__((ext_vector_type(8)))  _Float16 v8h;
typedef __attribute__((ext_vector_type(8)))  float    v8f;

#define BB 32
#define NREG 36
#define DD 2048
#define CC 4
#define HH 512
#define BN (BB * NREG)          // 1152
#define NPAD 48                 // 36 rows padded to 3 WMMA M-tiles

#define WMMA16(a, b, c) \
  __builtin_amdgcn_wmma_f32_16x16x32_f16(false, (a), false, (b), (short)0, (c), false, false)

__device__ __forceinline__ v16h cat8(v8h lo, v8h hi) {
  return __builtin_shufflevector(lo, hi, 0,1,2,3,4,5,6,7,8,9,10,11,12,13,14,15);
}

// ---------------- prep kernels ----------------

// mm (f32) -> mmh (f16), exact-size grid
__global__ __launch_bounds__(256) void cvt_mm_f16(const float* __restrict__ src,
                                                  _Float16* __restrict__ dst) {
  int i = blockIdx.x * 256 + threadIdx.x;
  dst[i] = (_Float16)src[i];
}

// transpose f32 [R,C] -> f16 [C,R]
__global__ __launch_bounds__(256) void transpose_w(const float* __restrict__ src,
                                                   _Float16* __restrict__ dst,
                                                   int R, int Ccols) {
  int i = blockIdx.x * 256 + threadIdx.x;   // index into dst, i = c*R + r
  int c = i / R;
  int r = i - c * R;
  dst[i] = (_Float16)src[r * Ccols + c];
}

// cl = coords @ W_cl ; cr = coords @ W_cr   (K = 4, scalar f32, store f16)
__global__ __launch_bounds__(256) void proj_coords(const float* __restrict__ coords,
                                                   const float* __restrict__ wcl,
                                                   const float* __restrict__ wcr,
                                                   _Float16* __restrict__ clh,
                                                   _Float16* __restrict__ crh) {
  int idx = blockIdx.x * 256 + threadIdx.x;   // < BN*HH
  int bn = idx >> 9;
  int h  = idx & (HH - 1);
  const float* c = coords + bn * CC;
  float al = 0.f, ar = 0.f;
#pragma unroll
  for (int k = 0; k < CC; ++k) {
    al += c[k] * wcl[k * HH + h];
    ar += c[k] * wcr[k * HH + h];
  }
  clh[idx] = (_Float16)al;
  crh[idx] = (_Float16)ar;
}

// fl = mm @ W_fl ; fr = mm @ W_fr  via WMMA.  A = mmh [BN, DD], Bt = W_t [HH, DD].
// One 16x16 output tile per wave. 4608 waves total (2304 tiles x 2 paths).
__global__ __launch_bounds__(256) void proj_feat(const _Float16* __restrict__ mmh,
                                                 const _Float16* __restrict__ wflt,
                                                 const _Float16* __restrict__ wfrt,
                                                 _Float16* __restrict__ flh,
                                                 _Float16* __restrict__ frh) {
  int gw   = blockIdx.x * 8 + (threadIdx.x >> 5);   // 0..4607
  int lane = threadIdx.x & 31;
  int path = (gw >= (BN / 16) * (HH / 16)) ? 1 : 0; // 2304 tiles per path
  int t    = gw - path * (BN / 16) * (HH / 16);
  int mt   = t >> 5;                                 // / (HH/16 = 32)
  int nt   = t & 31;

  const _Float16* bt  = path ? wfrt : wflt;
  _Float16*       dst = path ? frh  : flh;

  int r   = lane & 15;
  int klo = (lane < 16) ? 0 : 16;
  int kbo = (lane < 16) ? 0 : 16;
  const _Float16* arow = mmh + (mt * 16 + r) * DD + klo;
  const _Float16* brow = bt  + (nt * 16 + r) * DD + kbo;

  v8f acc = {};
  for (int kt = 0; kt < DD / 32; ++kt) {
    int k0 = kt * 32;
    v8h lo = *(const v8h*)(arow + k0);
    v8h hi = *(const v8h*)(arow + k0 + 8);
    v16h a = cat8(lo, hi);
    v16h b = *(const v16h*)(brow + k0);
    acc = WMMA16(a, b, acc);
  }

  int row0 = mt * 16 + ((lane < 16) ? 0 : 8);
  int col  = nt * 16 + r;
#pragma unroll
  for (int v = 0; v < 8; ++v) dst[(row0 + v) * HH + col] = (_Float16)acc[v];
}

// ---------------- main fused kernel ----------------
// One block per (b,i). Build P_c/P_f in LDS, GEMM against Wcout_t/Wfout_t [DD,HH],
// column-wise max over j, residual add.
__global__ __launch_bounds__(256) void pair_max_main(const float* __restrict__ mm,
                                                     const _Float16* __restrict__ clh,
                                                     const _Float16* __restrict__ crh,
                                                     const _Float16* __restrict__ flh,
                                                     const _Float16* __restrict__ frh,
                                                     const _Float16* __restrict__ wct,
                                                     const _Float16* __restrict__ wft,
                                                     float* __restrict__ out) {
  __shared__ __align__(32) _Float16 Pc[NPAD * HH];   // 48 KB
  __shared__ __align__(32) _Float16 Pf[NPAD * HH];   // 48 KB

  int bi   = blockIdx.x;          // 0..1151
  int b    = bi / NREG;
  int tid  = threadIdx.x;
  int lane = tid & 31;
  int wave = tid >> 5;

  // ---- fill P_c[j,h] = cl_i[h]*cr_j[h], P_f[j,h] = fl_i[h]*fr_j[h]; pad rows zero
  const _Float16* cl_i = clh + bi * HH;
  const _Float16* fl_i = flh + bi * HH;
  const _Float16* crB  = crh + b * NREG * HH;
  const _Float16* frB  = frh + b * NREG * HH;
  for (int c = tid; c < NPAD * (HH / 8); c += 256) {
    int j  = c >> 6;               // HH/8 = 64 chunks per row
    int h8 = (c & 63) * 8;
    v8h pc = {}, pf = {};
    if (j < NREG) {
      v8h cr = *(const v8h*)(crB + j * HH + h8);
      v8h fr = *(const v8h*)(frB + j * HH + h8);
      v8h cl = *(const v8h*)(cl_i + h8);
      v8h fl = *(const v8h*)(fl_i + h8);
      pc = cl * cr;
      pf = fl * fr;
    }
    *(v8h*)&Pc[j * HH + h8] = pc;
    *(v8h*)&Pf[j * HH + h8] = pf;
  }
  __syncthreads();

  int r   = lane & 15;
  int klo = (lane < 16) ? 0 : 16;
  int kbo = (lane < 16) ? 0 : 16;
  const float* mrow = mm  + bi * DD;
  float*       orow = out + bi * DD;

  // 128 N-tiles of 16 columns; each wave does 2 tiles per pass, 8 passes.
  for (int p = 0; p < 8; ++p) {
    int nt0 = p * 16 + wave * 2;
    v8f acc[2][3];
#pragma unroll
    for (int t = 0; t < 2; ++t)
#pragma unroll
      for (int m = 0; m < 3; ++m) acc[t][m] = (v8f){};

    const _Float16* bc0 = wct + ((nt0 + 0) * 16 + r) * HH + kbo;
    const _Float16* bc1 = wct + ((nt0 + 1) * 16 + r) * HH + kbo;
    const _Float16* bf0 = wft + ((nt0 + 0) * 16 + r) * HH + kbo;
    const _Float16* bf1 = wft + ((nt0 + 1) * 16 + r) * HH + kbo;

    for (int kt = 0; kt < HH / 32; ++kt) {
      int k0 = kt * 32;
      v16h a_c[3], a_f[3];
#pragma unroll
      for (int m = 0; m < 3; ++m) {
        int row = m * 16 + r;
        v8h lo  = *(const v8h*)&Pc[row * HH + k0 + klo];
        v8h hi  = *(const v8h*)&Pc[row * HH + k0 + klo + 8];
        a_c[m]  = cat8(lo, hi);
        v8h lo2 = *(const v8h*)&Pf[row * HH + k0 + klo];
        v8h hi2 = *(const v8h*)&Pf[row * HH + k0 + klo + 8];
        a_f[m]  = cat8(lo2, hi2);
      }
      v16h b_c0 = *(const v16h*)(bc0 + k0);
      v16h b_c1 = *(const v16h*)(bc1 + k0);
      v16h b_f0 = *(const v16h*)(bf0 + k0);
      v16h b_f1 = *(const v16h*)(bf1 + k0);
#pragma unroll
      for (int m = 0; m < 3; ++m) {
        acc[0][m] = WMMA16(a_c[m], b_c0, acc[0][m]);
        acc[1][m] = WMMA16(a_c[m], b_c1, acc[1][m]);
        acc[0][m] = WMMA16(a_f[m], b_f0, acc[0][m]);
        acc[1][m] = WMMA16(a_f[m], b_f1, acc[1][m]);
      }
    }

    // max over j. C layout: VGPR v, lanes 0-15 -> M = mtile*16 + v; lanes 16-31 -> +8.
    // Valid j < 36: tile2 only v<=3 on lanes<16.
#pragma unroll
    for (int t = 0; t < 2; ++t) {
      float mx = -3.402823466e38f;
#pragma unroll
      for (int v = 0; v < 8; ++v) {
        mx = fmaxf(mx, acc[t][0][v]);
        mx = fmaxf(mx, acc[t][1][v]);
      }
      if (lane < 16) {
#pragma unroll
        for (int v = 0; v < 4; ++v) mx = fmaxf(mx, acc[t][2][v]);
      }
      float o = __shfl_xor(mx, 16);
      mx = fmaxf(mx, o);
      if (lane < 16) {
        int d = (nt0 + t) * 16 + r;
        orow[d] = mx + mrow[d];
      }
    }
  }
}

// ---------------- launch ----------------
extern "C" void kernel_launch(void* const* d_in, const int* in_sizes, int n_in,
                              void* d_out, int out_size, void* d_ws, size_t ws_size,
                              hipStream_t stream) {
  const float* mm     = (const float*)d_in[0];
  const float* coords = (const float*)d_in[1];
  const float* W_cl   = (const float*)d_in[2];
  const float* W_cr   = (const float*)d_in[3];
  const float* W_cout = (const float*)d_in[4];
  const float* W_fl   = (const float*)d_in[5];
  const float* W_fr   = (const float*)d_in[6];
  const float* W_fout = (const float*)d_in[7];
  float* out = (float*)d_out;

  // workspace layout (all f16)
  _Float16* wsp  = (_Float16*)d_ws;
  _Float16* mmh  = wsp;                       // BN*DD   = 2359296
  _Float16* clh  = mmh  + (size_t)BN * DD;    // BN*HH   = 589824
  _Float16* crh  = clh  + (size_t)BN * HH;
  _Float16* flh  = crh  + (size_t)BN * HH;
  _Float16* frh  = flh  + (size_t)BN * HH;
  _Float16* wct  = frh  + (size_t)BN * HH;    // DD*HH   = 1048576
  _Float16* wft  = wct  + (size_t)DD * HH;
  _Float16* wflt = wft  + (size_t)DD * HH;
  _Float16* wfrt = wflt + (size_t)DD * HH;    // total ~17.8 MB

  // 1) conversions / transposes / coord projections
  cvt_mm_f16<<<(BN * DD) / 256, 256, 0, stream>>>(mm, mmh);
  transpose_w<<<(DD * HH) / 256, 256, 0, stream>>>(W_cout, wct, HH, DD);  // -> [DD,HH]
  transpose_w<<<(DD * HH) / 256, 256, 0, stream>>>(W_fout, wft, HH, DD);  // -> [DD,HH]
  transpose_w<<<(DD * HH) / 256, 256, 0, stream>>>(W_fl,  wflt, DD, HH);  // -> [HH,DD]
  transpose_w<<<(DD * HH) / 256, 256, 0, stream>>>(W_fr,  wfrt, DD, HH);  // -> [HH,DD]
  proj_coords<<<(BN * HH) / 256, 256, 0, stream>>>(coords, W_cl, W_cr, clh, crh);

  // 2) feature projections via WMMA: 4608 waves = 576 blocks x 8 waves
  proj_feat<<<576, 256, 0, stream>>>(mmh, wflt, wfrt, flh, frh);

  // 3) fused pairwise-bilinear + max + residual: one block per (b,i)
  pair_max_main<<<BN, 256, 0, stream>>>(mm, clh, crh, flh, frh, wct, wft, out);
}
